// MultiHeadLayer_35734127902891
// MI455X (gfx1250) — matrix operand order repaired and checked by reading, not compiled, not run on hardware
//
#include <hip/hip_runtime.h>

// ---------------------------------------------------------------------------
// Multi-head "interacting" attention (AutoInt style), full cross-head softmax.
//   B=4096, E=256, H=4, d=64.  Flash attention over all 16384 head-rows,
//   split-K over keys for occupancy, bf16 WMMA with f32 accumulation.
// ---------------------------------------------------------------------------

typedef __attribute__((ext_vector_type(16))) __bf16 v16bf;
typedef __attribute__((ext_vector_type(8)))  __bf16 v8bf;
typedef __attribute__((ext_vector_type(8)))  float  v8f;
typedef __attribute__((ext_vector_type(4)))  int    v4i;

union BF16x16 { v16bf v; v8bf h[2]; unsigned short u[16]; };

#define BATCH 4096
#define EMBED 256
#define HEADS 4
#define DHEAD 64
#define ROWS  (HEADS * BATCH)   // 16384
#define OUTW  (HEADS * DHEAD)   // 256
#define KT    64                // key-tile size
#define LPAD  72                // padded LDS row stride (elements)

__device__ __forceinline__ unsigned short f2bf(float f) {
    unsigned int u = __float_as_uint(f);
    u += 0x7FFFu + ((u >> 16) & 1u);          // round-to-nearest-even
    return (unsigned short)(u >> 16);
}

// ---- CDNA5 async global->LDS staging (guarded; falls back to plain copy) ---
#if defined(__HIP_DEVICE_COMPILE__) && defined(__gfx1250__) && \
    __has_builtin(__builtin_amdgcn_global_load_async_to_lds_b128)
#define HAS_ASYNC 1
#else
#define HAS_ASYNC 0
#endif

__device__ __forceinline__ void cp16_tile(const unsigned short* g, unsigned short* l) {
#if HAS_ASYNC
    auto gp = (__attribute__((address_space(1))) v4i*)(unsigned long long)(size_t)g;
    auto lp = (__attribute__((address_space(3))) v4i*)(unsigned int)(size_t)l;
    __builtin_amdgcn_global_load_async_to_lds_b128(gp, lp, 0, 0);
#else
    *(uint4*)l = *(const uint4*)g;
#endif
}

__device__ __forceinline__ void tile_copy_fence() {
#if HAS_ASYNC
#if __has_builtin(__builtin_amdgcn_s_wait_asynccnt)
    __builtin_amdgcn_s_wait_asynccnt(0);
#else
    asm volatile("s_wait_asynccnt 0x0" ::: "memory");
#endif
#endif
}

// ---------------------------------------------------------------------------
// Stage 1: C[4096x256] = X[4096x256] @ W[256x256] in bf16 WMMA.
//   VT=0 : head-split rows   Out[(head*BATCH + b)*64 + c]          (Q, K)
//   VT=1 : transposed store  Out[c*ROWS + head*BATCH + b]          (V^T)
// One 16x16 tile per wave, K=256 via eight chained v_wmma_f32_16x16x32_bf16.
// ---------------------------------------------------------------------------
template <int VT>
__global__ __launch_bounds__(256) void proj_kernel(const float* __restrict__ X,
                                                   const float* __restrict__ W,
                                                   unsigned short* __restrict__ Out) {
    const int lane = threadIdx.x & 31;
    const int ln16 = lane & 15;
    const bool hi  = (lane & 16) != 0;
    const int wid  = blockIdx.x * 8 + (threadIdx.x >> 5);
    const int tm   = wid >> 4;                 // 0..255 row tile
    const int tn   = wid & 15;                 // 0..15  col tile
    const int mrow = tm * 16 + ln16;           // A-lane row
    const int bn   = tn * 16 + ln16;           // B-lane col

    v8f acc = {0.f, 0.f, 0.f, 0.f, 0.f, 0.f, 0.f, 0.f};
    for (int kk = 0; kk < EMBED; kk += 32) {
        BF16x16 a, b;
        // A fragment: lane=m, K chunks {0-7,16-23} (lo half) / {8-15,24-31} (hi)
        const float* xr = X + mrow * EMBED + kk + (hi ? 8 : 0);
        #pragma unroll
        for (int j = 0; j < 8; ++j) {
            a.u[j]     = f2bf(xr[j]);
            a.u[j + 8] = f2bf(xr[j + 16]);
        }
        // B fragment: lane=n, K chunks {0-7,16-23} (lo half) / {8-15,24-31} (hi)
        const int kb = kk + (hi ? 8 : 0);
        #pragma unroll
        for (int j = 0; j < 8; ++j) {
            b.u[j]     = f2bf(W[(kb + j) * OUTW + bn]);
            b.u[j + 8] = f2bf(W[(kb + j + 16) * OUTW + bn]);
        }
        acc = __builtin_amdgcn_wmma_f32_16x16x32_bf16(false, a.v, false, b.v,
                                                      (short)0, acc, false, false);
    }

    #pragma unroll
    for (int r = 0; r < 8; ++r) {
        const int row  = tm * 16 + r + (hi ? 8 : 0);   // batch index b
        const int col  = tn * 16 + ln16;               // 0..255
        const int head = col >> 6;
        const int c    = col & 63;
        if (VT == 0)
            Out[(head * BATCH + row) * DHEAD + c] = f2bf(acc[r]);
        else
            Out[(size_t)c * ROWS + head * BATCH + row] = f2bf(acc[r]);
    }
}

// ---------------------------------------------------------------------------
// Stage 2: flash attention over a key chunk.  Block = 4 waves (128 thr),
// each wave owns 16 query rows; blockIdx.y selects the key split.
// finalize=1 (single split): normalize and write merged output directly.
// finalize=0: write unnormalized O plus per-row (m, l) partials.
// ---------------------------------------------------------------------------
__global__ __launch_bounds__(128) void attn_kernel(const unsigned short* __restrict__ Qh,
                                                   const unsigned short* __restrict__ Kh,
                                                   const unsigned short* __restrict__ Vt,
                                                   float* __restrict__ out,
                                                   float* __restrict__ Opart,
                                                   float* __restrict__ Mpart,
                                                   float* __restrict__ Lpart,
                                                   int chunkRows, int finalize) {
    __shared__ __align__(16) unsigned short Ksh [KT * LPAD];       // K tile row-major
    __shared__ __align__(16) unsigned short Vtsh[DHEAD * LPAD];    // V^T tile row-major
    __shared__ __align__(16) unsigned short Ssh [4 * 16 * LPAD];   // per-wave P scratch

    const int tid   = threadIdx.x;
    const int lane  = tid & 31;
    const int wave  = tid >> 5;
    const int ln16  = lane & 15;
    const bool hi   = (lane & 16) != 0;
    const int qrow0 = blockIdx.x * 64 + wave * 16;

    // --- Q fragments (row = qrow0 + ln16), reused across all key tiles ---
    BF16x16 aq[2];
    {
        const unsigned short* qr = Qh + (qrow0 + ln16) * DHEAD;
        #pragma unroll
        for (int f = 0; f < 2; ++f) {
            const int K0 = f * 32 + (hi ? 8 : 0);
            aq[f].h[0] = *(const v8bf*)(qr + K0);
            aq[f].h[1] = *(const v8bf*)(qr + K0 + 16);
        }
    }

    float mrun[8], lrun[8];
    v8f o[4];
    #pragma unroll
    for (int r = 0; r < 8; ++r) { mrun[r] = -1e30f; lrun[r] = 0.0f; }
    #pragma unroll
    for (int n = 0; n < 4; ++n) o[n] = (v8f){0.f,0.f,0.f,0.f,0.f,0.f,0.f,0.f};

    unsigned short* sw = Ssh + wave * 16 * LPAD;
    const int kb0 = blockIdx.y * chunkRows;

    for (int kb = kb0; kb < kb0 + chunkRows; kb += KT) {
        __syncthreads();
        // ---- stage K (64x64) and V^T (64x64) tiles: 8 b128 copies/thread ----
        #pragma unroll
        for (int j = 0; j < 4; ++j) {
            const int i   = tid + j * 128;     // 0..511
            const int row = i >> 3;
            const int col = (i & 7) * 8;
            cp16_tile(Kh + (size_t)(kb + row) * DHEAD + col, Ksh  + row * LPAD + col);
            cp16_tile(Vt + (size_t)row * ROWS + kb + col,    Vtsh + row * LPAD + col);
        }
        tile_copy_fence();
        __syncthreads();

        // ---- S = (Q K^T) * 1/sqrt(d): 8 WMMAs ----
        v8f s[4];
        #pragma unroll
        for (int nt = 0; nt < 4; ++nt) {
            v8f c = {0.f,0.f,0.f,0.f,0.f,0.f,0.f,0.f};
            #pragma unroll
            for (int kf = 0; kf < 2; ++kf) {
                BF16x16 bk;  // B[k,n] = K[n,k] -> lane=n reads row n of K tile
                const unsigned short* kr2 = Ksh + (nt * 16 + ln16) * LPAD
                                                + kf * 32 + (hi ? 8 : 0);
                bk.h[0] = *(const v8bf*)(kr2);
                bk.h[1] = *(const v8bf*)(kr2 + 16);
                c = __builtin_amdgcn_wmma_f32_16x16x32_bf16(false, aq[kf].v, false, bk.v,
                                                            (short)0, c, false, false);
            }
            s[nt] = c * 0.125f;
        }

        // ---- online softmax (row-wise over the 16-lane halves) ----
        float scl[8];
        #pragma unroll
        for (int r = 0; r < 8; ++r) {
            float v = fmaxf(fmaxf(s[0][r], s[1][r]), fmaxf(s[2][r], s[3][r]));
            v = fmaxf(v, __shfl_xor(v, 1, 32));
            v = fmaxf(v, __shfl_xor(v, 2, 32));
            v = fmaxf(v, __shfl_xor(v, 4, 32));
            v = fmaxf(v, __shfl_xor(v, 8, 32));
            const float mn = fmaxf(mrun[r], v);
            scl[r]  = __expf(mrun[r] - mn);
            mrun[r] = mn;
            #pragma unroll
            for (int nt = 0; nt < 4; ++nt)
                s[nt][r] = __expf(s[nt][r] - mn);
            float sum = s[0][r] + s[1][r] + s[2][r] + s[3][r];
            sum += __shfl_xor(sum, 1, 32);
            sum += __shfl_xor(sum, 2, 32);
            sum += __shfl_xor(sum, 4, 32);
            sum += __shfl_xor(sum, 8, 32);
            lrun[r] = lrun[r] * scl[r] + sum;
        }

        // ---- C-layout -> A-layout for P via per-wave LDS scratch ----
        #pragma unroll
        for (int nt = 0; nt < 4; ++nt)
            #pragma unroll
            for (int r = 0; r < 8; ++r)
                sw[(r + (hi ? 8 : 0)) * LPAD + nt * 16 + ln16] = f2bf(s[nt][r]);

        BF16x16 ap[2];
        {
            const unsigned short* pr = sw + ln16 * LPAD;
            #pragma unroll
            for (int f = 0; f < 2; ++f) {
                const int K0 = f * 32 + (hi ? 8 : 0);
                ap[f].h[0] = *(const v8bf*)(pr + K0);
                ap[f].h[1] = *(const v8bf*)(pr + K0 + 16);
            }
        }

        // ---- O = O*scale + P @ V: 8 WMMAs ----
        #pragma unroll
        for (int nt = 0; nt < 4; ++nt) {
            #pragma unroll
            for (int r = 0; r < 8; ++r) o[nt][r] *= scl[r];
            #pragma unroll
            for (int kf = 0; kf < 2; ++kf) {
                BF16x16 bv;  // B[k,n] = V[k,n] -> lane=n reads row n of V^T tile
                const unsigned short* vr = Vtsh + (nt * 16 + ln16) * LPAD
                                                + kf * 32 + (hi ? 8 : 0);
                bv.h[0] = *(const v8bf*)(vr);
                bv.h[1] = *(const v8bf*)(vr + 16);
                o[nt] = __builtin_amdgcn_wmma_f32_16x16x32_bf16(false, ap[kf].v, false, bv.v,
                                                                (short)0, o[nt], false, false);
            }
        }
    }

    if (finalize) {
        // ---- single split: normalize + write merged-head fp32 output ----
        #pragma unroll
        for (int r = 0; r < 8; ++r) {
            const int g    = qrow0 + r + (hi ? 8 : 0);
            const int head = g >> 12;
            const int b    = g & (BATCH - 1);
            const float inv = 1.0f / lrun[r];
            #pragma unroll
            for (int nt = 0; nt < 4; ++nt)
                out[b * OUTW + head * DHEAD + nt * 16 + ln16] = o[nt][r] * inv;
        }
    } else {
        // ---- write unnormalized partials for the combine pass ----
        const int sp = blockIdx.y;
        float* Og = Opart + (size_t)sp * ROWS * DHEAD;
        #pragma unroll
        for (int r = 0; r < 8; ++r) {
            const int g = qrow0 + r + (hi ? 8 : 0);
            #pragma unroll
            for (int nt = 0; nt < 4; ++nt)
                Og[(size_t)g * DHEAD + nt * 16 + ln16] = o[nt][r];
            if (ln16 == 0) {
                Mpart[(size_t)sp * ROWS + g] = mrun[r];
                Lpart[(size_t)sp * ROWS + g] = lrun[r];
            }
        }
    }
}

// ---------------------------------------------------------------------------
// Stage 3: flash-combine the key-split partials and write merged output.
// ---------------------------------------------------------------------------
__global__ __launch_bounds__(256) void combine_kernel(const float* __restrict__ Opart,
                                                      const float* __restrict__ Mpart,
                                                      const float* __restrict__ Lpart,
                                                      float* __restrict__ out, int split) {
    const int idx = blockIdx.x * 256 + threadIdx.x;    // 0 .. ROWS*DHEAD-1
    const int g = idx >> 6;
    const int c = idx & 63;
    float mmax = -1e30f;
    for (int s = 0; s < split; ++s)
        mmax = fmaxf(mmax, Mpart[(size_t)s * ROWS + g]);
    float num = 0.f, den = 0.f;
    for (int s = 0; s < split; ++s) {
        const float w = __expf(Mpart[(size_t)s * ROWS + g] - mmax);
        num += w * Opart[((size_t)s * ROWS + g) * DHEAD + c];
        den += w * Lpart[(size_t)s * ROWS + g];
    }
    const int head = g >> 12;
    const int b    = g & (BATCH - 1);
    out[b * OUTW + head * DHEAD + c] = num / den;
}

// ---------------------------------------------------------------------------
extern "C" void kernel_launch(void* const* d_in, const int* in_sizes, int n_in,
                              void* d_out, int out_size, void* d_ws, size_t ws_size,
                              hipStream_t stream) {
    const float* queries = (const float*)d_in[0];
    const float* keys    = (const float*)d_in[1];
    const float* values  = (const float*)d_in[2];
    const float* Wq      = (const float*)d_in[3];
    const float* Wk      = (const float*)d_in[4];
    const float* Wv      = (const float*)d_in[5];
    float* out = (float*)d_out;

    unsigned short* qh = (unsigned short*)d_ws;            // [16384][64] bf16
    unsigned short* kh = qh + (size_t)ROWS * DHEAD;        // [16384][64] bf16
    unsigned short* vt = kh + (size_t)ROWS * DHEAD;        // [64][16384] bf16 (V^T)

    const size_t qkv_bytes = 3ull * ROWS * DHEAD * sizeof(unsigned short);   // 6 MB
    auto need = [&](int s) {
        return qkv_bytes + (size_t)s * ROWS * (DHEAD * sizeof(float) + 2 * sizeof(float));
    };
    int split = 1;
    if (ws_size >= need(4))      split = 4;
    else if (ws_size >= need(2)) split = 2;

    float* Opart = (float*)((char*)d_ws + qkv_bytes);
    float* Mpart = Opart + (size_t)split * ROWS * DHEAD;
    float* Lpart = Mpart + (size_t)split * ROWS;

    // Stage 1: projections (Q,K head-split; V transposed), 512 blocks each.
    proj_kernel<0><<<dim3(512), dim3(256), 0, stream>>>(queries, Wq, qh);
    proj_kernel<0><<<dim3(512), dim3(256), 0, stream>>>(keys,    Wk, kh);
    proj_kernel<1><<<dim3(512), dim3(256), 0, stream>>>(values,  Wv, vt);

    // Stage 2 (+3): flash attention, 64 query rows per block, split-K over keys.
    if (split == 1) {
        attn_kernel<<<dim3(ROWS / 64, 1), dim3(128), 0, stream>>>(
            qh, kh, vt, out, nullptr, nullptr, nullptr, ROWS, 1);
    } else {
        attn_kernel<<<dim3(ROWS / 64, split), dim3(128), 0, stream>>>(
            qh, kh, vt, out, Opart, Mpart, Lpart, ROWS / split, 0);
        combine_kernel<<<dim3(ROWS * DHEAD / 256), dim3(256), 0, stream>>>(
            Opart, Mpart, Lpart, out, split);
    }
}
